// AdaptiveSparseAttention_34411277976178
// MI455X (gfx1250) — matrix operand order repaired and checked by reading, not compiled, not run on hardware
//
#include <hip/hip_runtime.h>
#include <hip/hip_bf16.h>

// Problem constants (B,L,D,H) = (4,1024,512,8), head_dim=64
#define B_    4
#define L_    1024
#define D_    512
#define H_    8
#define HD_   64
#define KTOP  716          // int(1024 * (1 - 0.3))
#define THRESH 0.05f

typedef __attribute__((ext_vector_type(16))) __bf16 v16bf;
typedef __attribute__((ext_vector_type(8)))  float  v8f;

union FragB {
  v16bf v;
  unsigned short u[16];
  uint4 q[2];
};

__device__ __forceinline__ unsigned short f2bf(float f) {
  unsigned int u = __float_as_uint(f);
  u += 0x7FFFu + ((u >> 16) & 1u);         // round-to-nearest-even
  return (unsigned short)(u >> 16);
}

__device__ __forceinline__ v8f wmma_bf16(v16bf a, v16bf b, v8f c) {
  // v_wmma_f32_16x16x32_bf16  (neg_a, A, neg_b, B, c_mod, C, reuse_a, reuse_b)
  return __builtin_amdgcn_wmma_f32_16x16x32_bf16(false, a, false, b, (short)0, c,
                                                 false, false);
}

// A-matrix fragment 16x32 bf16, CDNA5 wave32 layout:
// lane = m + 16*(k_half/8 group); per lane two contiguous 8-elem chunks at
// k = g*8 and k = 16 + g*8  (g = lane>>4).  Works for LDS or global pointers.
__device__ __forceinline__ v16bf lda16x32(const unsigned short* s, int row0,
                                          int stride, int lane) {
  FragB f;
  int m  = row0 + (lane & 15);
  int kg = (lane >> 4) * 8;
  const unsigned short* p = s + m * stride + kg;
  f.q[0] = *(const uint4*)(p);
  f.q[1] = *(const uint4*)(p + 16);
  return f.v;
}

// B-matrix fragment 32x16 bf16 from n-major storage s[n][stride]:
// lane = n + 16*g; per lane two 8-elem chunks at k = g*8 and k = 16 + g*8.
__device__ __forceinline__ v16bf ldb32x16(const unsigned short* s, int n0,
                                          int stride, int lane) {
  FragB f;
  int n  = n0 + (lane & 15);
  int kb = (lane >> 4) * 8;
  const unsigned short* p = s + n * stride + kb;
  f.q[0] = *(const uint4*)(p);
  f.q[1] = *(const uint4*)(p + 16);
  return f.v;
}

// ---------------------------------------------------------------------------
// Kernel 1: pattern-selector MLP -> softmax weights pw[B][3]
// ---------------------------------------------------------------------------
__global__ __launch_bounds__(256) void pattern_kernel(
    const float* __restrict__ x,
    const float* __restrict__ w1, const float* __restrict__ b1,
    const float* __restrict__ w2, const float* __restrict__ b2,
    const float* __restrict__ w3, const float* __restrict__ b3,
    const float* __restrict__ pbias, float* __restrict__ pw) {
  __shared__ float pooled[D_];
  __shared__ float h1[D_];
  __shared__ float h2[D_ / 2];
  __shared__ float lg[4];
  int b = blockIdx.x, tid = threadIdx.x;

  for (int d = tid; d < D_; d += 256) {
    float s = 0.f;
    for (int l = 0; l < L_; ++l) s += x[((long)b * L_ + l) * D_ + d];
    pooled[d] = s * (1.0f / (float)L_);
  }
  __syncthreads();
  for (int o = tid; o < D_; o += 256) {
    float s = b1[o];
    for (int i = 0; i < D_; ++i) s += pooled[i] * w1[o * D_ + i];
    h1[o] = fmaxf(s, 0.f);
  }
  __syncthreads();
  for (int o = tid; o < D_ / 2; o += 256) {
    float s = b2[o];
    for (int i = 0; i < D_; ++i) s += h1[i] * w2[o * D_ + i];
    h2[o] = fmaxf(s, 0.f);
  }
  __syncthreads();
  if (tid < 3) {
    float s = b3[tid] + pbias[tid];
    for (int i = 0; i < D_ / 2; ++i) s += h2[i] * w3[tid * (D_ / 2) + i];
    lg[tid] = s;
  }
  __syncthreads();
  if (tid == 0) {
    float m = fmaxf(lg[0], fmaxf(lg[1], lg[2]));
    float e0 = expf((lg[0] - m) * (1.0f / 0.3f));
    float e1 = expf((lg[1] - m) * (1.0f / 0.3f));
    float e2 = expf((lg[2] - m) * (1.0f / 0.3f));
    float inv = 1.0f / (e0 + e1 + e2);
    pw[b * 3 + 0] = e0 * inv;
    pw[b * 3 + 1] = e1 * inv;
    pw[b * 3 + 2] = e2 * inv;
  }
}

// ---------------------------------------------------------------------------
// Kernel 2: QKV GEMM  (4096x1536) = X(4096x512) @ qkv_w^T, scatter to
// q/k/v bf16 arrays laid out (B,H,L,64).  Block tile 128x64, 8 waves (2x2
// 16x16 WMMA tiles per wave), K staged in LDS as bf16.
// ---------------------------------------------------------------------------
__global__ __launch_bounds__(256) void qkv_kernel(
    const float* __restrict__ X, const float* __restrict__ W,
    unsigned short* __restrict__ qb, unsigned short* __restrict__ kb,
    unsigned short* __restrict__ vb) {
  __shared__ __align__(16) unsigned short sA[128 * 32];
  __shared__ __align__(16) unsigned short sB[64 * 32];
  int tid = threadIdx.x, lane = tid & 31, wave = tid >> 5;
  int wm = (wave & 3) * 32, wn = (wave >> 2) * 32;
  int m0 = blockIdx.x * 128, n0 = blockIdx.y * 64;

  v8f acc[2][2];
  for (int mi = 0; mi < 2; ++mi)
    for (int ni = 0; ni < 2; ++ni)
      for (int r = 0; r < 8; ++r) acc[mi][ni][r] = 0.f;

  for (int kk = 0; kk < D_; kk += 32) {
    for (int idx = tid; idx < 128 * 32; idx += 256) {
      int r = idx >> 5, c = idx & 31;
      sA[idx] = f2bf(X[(long)(m0 + r) * D_ + kk + c]);
      if (kk + 32 < D_) __builtin_prefetch(&X[(long)(m0 + r) * D_ + kk + 32 + c], 0, 1);
    }
    for (int idx = tid; idx < 64 * 32; idx += 256) {
      int r = idx >> 5, c = idx & 31;
      sB[idx] = f2bf(W[(long)(n0 + r) * D_ + kk + c]);
    }
    __syncthreads();
    v16bf a0 = lda16x32(sA, wm, 32, lane);
    v16bf a1 = lda16x32(sA, wm + 16, 32, lane);
    for (int ni = 0; ni < 2; ++ni) {
      v16bf bf = ldb32x16(sB, wn + ni * 16, 32, lane);
      acc[0][ni] = wmma_bf16(a0, bf, acc[0][ni]);
      acc[1][ni] = wmma_bf16(a1, bf, acc[1][ni]);
    }
    __syncthreads();
  }

  for (int mi = 0; mi < 2; ++mi)
    for (int ni = 0; ni < 2; ++ni)
      for (int r = 0; r < 8; ++r) {
        int gm = m0 + wm + mi * 16 + r + 8 * (lane >> 4);
        int o  = n0 + wn + ni * 16 + (lane & 15);
        int which = o >> 9, d = o & 511;
        int h = d >> 6, hd = d & 63;
        int bb = gm >> 10, l = gm & 1023;
        long di = (((long)(bb * H_ + h)) * L_ + l) * HD_ + hd;
        unsigned short val = f2bf(acc[mi][ni][r]);
        if (which == 0) qb[di] = val;
        else if (which == 1) kb[di] = val;
        else vb[di] = val;
      }
}

// ---------------------------------------------------------------------------
// Kernel 3: fused scores + top-k threshold + mask blend + softmax + attn@V
// One block per (b, h, 16-query-row tile); 128 threads = 4 waves.
// Dynamic LDS: 16x1024 f32 scores (64KB) + 64x32 bf16 V stage + reductions.
// ---------------------------------------------------------------------------
__global__ __launch_bounds__(128) void attn_kernel(
    const unsigned short* __restrict__ qb, const unsigned short* __restrict__ kb,
    const unsigned short* __restrict__ vb, const float* __restrict__ pwbuf,
    const float* __restrict__ sparse_w, const float* __restrict__ sparse_b,
    unsigned short* __restrict__ ao) {
  extern __shared__ char smem[];
  float* sc           = (float*)smem;                       // 16*1024 f32
  unsigned short* sv  = (unsigned short*)(sc + 16 * 1024);  // 64*32 bf16
  float* red          = (float*)(sv + 64 * 32);             // 128
  float* stat         = red + 128;                          // 16

  int tid = threadIdx.x, lane = tid & 31, wave = tid >> 5;
  int qt = blockIdx.x, h = blockIdx.y, b = blockIdx.z;
  long base = ((long)(b * H_ + h)) * L_ * HD_;
  const unsigned short* Q = qb + base;
  const unsigned short* K = kb + base;
  const unsigned short* V = vb + base;
  int q0 = qt * 16;

  // ---- phase 1: raw scores (16 x 1024) via WMMA, scaled by hd^-0.5 ----
  v16bf a0 = lda16x32(Q, q0, HD_, lane);        // k-dims 0..31
  v16bf a1 = lda16x32(Q + 32, q0, HD_, lane);   // k-dims 32..63
  for (int ct = wave; ct < L_ / 16; ct += 4) {
    int n0 = ct * 16;
    v16bf b0 = ldb32x16(K, n0, HD_, lane);
    v16bf b1 = ldb32x16(K + 32, n0, HD_, lane);
    v8f acc;
    for (int r = 0; r < 8; ++r) acc[r] = 0.f;
    acc = wmma_bf16(a0, b0, acc);
    acc = wmma_bf16(a1, b1, acc);
    for (int r = 0; r < 8; ++r) {
      int row = r + 8 * (lane >> 4);
      int col = n0 + (lane & 15);
      sc[row * L_ + col] = acc[r] * 0.125f;     // 64^-0.5
    }
  }
  __syncthreads();

  // ---- phase 2: top-k threshold + mask blend + softmax (8 threads/row) ----
  float pw0 = pwbuf[b * 3 + 0], pw1 = pwbuf[b * 3 + 1], pw2 = pwbuf[b * 3 + 2];
  float sw = sparse_w[h], sb = sparse_b[h];
  bool t00 = pw1 > THRESH;
  bool t10 = (pw0 + pw1) > THRESH;
  bool t01 = (pw1 + pw2) > THRESH;
  bool t11 = (pw0 + pw1 + pw2) > THRESH;

  int row = tid >> 3, sub = tid & 7;
  float* rs = sc + row * L_;
  int j0 = sub * 128, j1 = j0 + 128;

  float mn = 3e38f, mx = -3e38f;
  for (int j = j0; j < j1; ++j) {
    float s2 = rs[j] * sw + sb;
    mn = fminf(mn, s2);
    mx = fmaxf(mx, s2);
  }
  red[tid] = mx; __syncthreads();
  if (sub == 0) {
    float v = red[row * 8];
    for (int i = 1; i < 8; ++i) v = fmaxf(v, red[row * 8 + i]);
    stat[row] = v;
  }
  __syncthreads();
  float hi = stat[row] + 1e-3f;
  red[tid] = mn; __syncthreads();
  if (sub == 0) {
    float v = red[row * 8];
    for (int i = 1; i < 8; ++i) v = fminf(v, red[row * 8 + i]);
    stat[row] = v;
  }
  __syncthreads();
  float lo = stat[row] - 1e-3f;

  // bisection: theta s.t. count(s2 >= theta) ~= KTOP (monotone in theta)
  for (int it = 0; it < 28; ++it) {
    float mid = 0.5f * (lo + hi);
    float c = 0.f;
    for (int j = j0; j < j1; ++j)
      c += (rs[j] * sw + sb >= mid) ? 1.f : 0.f;
    red[tid] = c; __syncthreads();
    if (sub == 0) {
      float s = 0.f;
      for (int i = 0; i < 8; ++i) s += red[row * 8 + i];
      stat[row] = s;
    }
    __syncthreads();
    float cnt = stat[row];
    if (cnt >= (float)KTOP) lo = mid; else hi = mid;
  }
  float theta = lo;

  // mask blend (combined = pw0*local + pw1 + pw2*sparse > 0.05), row max, count
  int qi = q0 + row;
  float m = -3e38f, ca = 0.f;
  for (int j = j0; j < j1; ++j) {
    float v = rs[j];
    bool loc = (qi - j <= 16) && (j - qi <= 16);
    bool sp  = (v * sw + sb) >= theta;
    bool allow = loc ? (sp ? t11 : t10) : (sp ? t01 : t00);
    float nv = allow ? v : -1e30f;
    rs[j] = nv;
    if (allow) { ca += 1.f; m = fmaxf(m, nv); }
  }
  red[tid] = ca; __syncthreads();
  if (sub == 0) {
    float s = 0.f;
    for (int i = 0; i < 8; ++i) s += red[row * 8 + i];
    stat[row] = s;
  }
  __syncthreads();
  float cAll = stat[row];
  red[tid] = m; __syncthreads();
  if (sub == 0) {
    float v = red[row * 8];
    for (int i = 1; i < 8; ++i) v = fmaxf(v, red[row * 8 + i]);
    stat[row] = v;
  }
  __syncthreads();
  float rmax = stat[row];
  if (cAll < 0.5f) {           // fully-masked row: unmask position 0 with 0.0
    if (sub == 0) rs[0] = 0.f;
    rmax = 0.f;
  }
  __syncthreads();

  float es = 0.f;
  for (int j = j0; j < j1; ++j) {
    float e = expf(rs[j] - rmax);   // masked (-1e30) underflows to 0
    rs[j] = e;
    es += e;
  }
  red[tid] = es; __syncthreads();
  if (sub == 0) {
    float s = 0.f;
    for (int i = 0; i < 8; ++i) s += red[row * 8 + i];
    stat[row] = s;
  }
  __syncthreads();
  float inv = 1.0f / stat[row];
  for (int j = j0; j < j1; ++j) rs[j] *= inv;
  __syncthreads();

  // ---- phase 3: out = probs(16x1024) @ V(1024x64) via WMMA ----
  v8f oacc;
  for (int r = 0; r < 8; ++r) oacc[r] = 0.f;
  int nw = wave * 16;                 // this wave's 16 head-dim columns
  for (int kt = 0; kt < L_ / 32; ++kt) {
    int k0 = kt * 32;
    __syncthreads();                  // protect sv reads from prev iter
    for (int idx = tid; idx < 64 * 32; idx += 128) {
      int kk2 = idx >> 6, hd = idx & 63;
      sv[hd * 32 + kk2] = V[(long)(k0 + kk2) * HD_ + hd];  // transpose stage
    }
    __syncthreads();
    FragB af;                         // A frag from f32 probs in LDS
    int mrow = lane & 15;
    int kg = (lane >> 4) * 8;
    for (int r = 0; r < 8; ++r) {
      int kloc = (r >> 2) * 16 + kg + ((r & 3) << 1);
      af.u[2 * r]     = f2bf(sc[mrow * L_ + k0 + kloc]);
      af.u[2 * r + 1] = f2bf(sc[mrow * L_ + k0 + kloc + 1]);
    }
    v16bf bf = ldb32x16(sv, nw, 32, lane);
    oacc = wmma_bf16(af.v, bf, oacc);
  }

  for (int r = 0; r < 8; ++r) {
    int l = q0 + r + 8 * (lane >> 4);
    int d = h * 64 + nw + (lane & 15);
    ao[((long)(b * L_ + l)) * D_ + d] = f2bf(oacc[r]);
  }
}

// ---------------------------------------------------------------------------
// Kernel 4: output projection (4096x512) = AO(4096x512, bf16) @ proj_w^T + b
// ---------------------------------------------------------------------------
__global__ __launch_bounds__(256) void proj_kernel(
    const unsigned short* __restrict__ AO, const float* __restrict__ W,
    const float* __restrict__ bias, float* __restrict__ out) {
  __shared__ __align__(16) unsigned short sA[128 * 32];
  __shared__ __align__(16) unsigned short sB[64 * 32];
  int tid = threadIdx.x, lane = tid & 31, wave = tid >> 5;
  int wm = (wave & 3) * 32, wn = (wave >> 2) * 32;
  int m0 = blockIdx.x * 128, n0 = blockIdx.y * 64;

  v8f acc[2][2];
  for (int mi = 0; mi < 2; ++mi)
    for (int ni = 0; ni < 2; ++ni)
      for (int r = 0; r < 8; ++r) acc[mi][ni][r] = 0.f;

  for (int kk = 0; kk < D_; kk += 32) {
    for (int idx = tid; idx < 128 * 32; idx += 256) {
      int r = idx >> 5, c = idx & 31;
      sA[idx] = AO[(long)(m0 + r) * D_ + kk + c];          // already bf16
    }
    for (int idx = tid; idx < 64 * 32; idx += 256) {
      int r = idx >> 5, c = idx & 31;
      sB[idx] = f2bf(W[(long)(n0 + r) * D_ + kk + c]);
    }
    __syncthreads();
    v16bf a0 = lda16x32(sA, wm, 32, lane);
    v16bf a1 = lda16x32(sA, wm + 16, 32, lane);
    for (int ni = 0; ni < 2; ++ni) {
      v16bf bf = ldb32x16(sB, wn + ni * 16, 32, lane);
      acc[0][ni] = wmma_bf16(a0, bf, acc[0][ni]);
      acc[1][ni] = wmma_bf16(a1, bf, acc[1][ni]);
    }
    __syncthreads();
  }

  for (int mi = 0; mi < 2; ++mi)
    for (int ni = 0; ni < 2; ++ni)
      for (int r = 0; r < 8; ++r) {
        int gm = m0 + wm + mi * 16 + r + 8 * (lane >> 4);
        int o  = n0 + wn + ni * 16 + (lane & 15);
        out[(long)gm * D_ + o] = acc[mi][ni][r] + bias[o];
      }
}

// ---------------------------------------------------------------------------
extern "C" void kernel_launch(void* const* d_in, const int* in_sizes, int n_in,
                              void* d_out, int out_size, void* d_ws, size_t ws_size,
                              hipStream_t stream) {
  const float* x            = (const float*)d_in[0];
  const float* qkv_w        = (const float*)d_in[1];
  const float* proj_w       = (const float*)d_in[2];
  const float* proj_b       = (const float*)d_in[3];
  const float* ps_w1        = (const float*)d_in[4];
  const float* ps_b1        = (const float*)d_in[5];
  const float* ps_w2        = (const float*)d_in[6];
  const float* ps_b2        = (const float*)d_in[7];
  const float* ps_w3        = (const float*)d_in[8];
  const float* ps_b3        = (const float*)d_in[9];
  const float* pattern_bias = (const float*)d_in[10];
  const float* sparse_w     = (const float*)d_in[11];
  const float* sparse_b     = (const float*)d_in[12];

  // workspace layout: pw[12] | q,k,v,attn_out bf16 (2M elements each)
  float* pw = (float*)d_ws;
  unsigned short* qb = (unsigned short*)((char*)d_ws + 256);
  size_t n1 = (size_t)B_ * H_ * L_ * HD_;     // 2M
  unsigned short* kb = qb + n1;
  unsigned short* vb = kb + n1;
  unsigned short* ao = vb + n1;

  pattern_kernel<<<B_, 256, 0, stream>>>(x, ps_w1, ps_b1, ps_w2, ps_b2,
                                         ps_w3, ps_b3, pattern_bias, pw);

  qkv_kernel<<<dim3(32, 24), 256, 0, stream>>>(x, qkv_w, qb, kb, vb);

  size_t smem = (size_t)16 * 1024 * 4 + 64 * 32 * 2 + (128 + 16) * 4 + 64;
  attn_kernel<<<dim3(L_ / 16, H_, B_), 128, smem, stream>>>(
      qb, kb, vb, pw, sparse_w, sparse_b, ao);

  proj_kernel<<<dim3(32, 8), 256, 0, stream>>>(ao, proj_w, proj_b,
                                               (float*)d_out);
}
